// ConformerAttention_76733885710494
// MI455X (gfx1250) — compile-verified, matrix operand order checked by reading, not run on hardware
//
#include <hip/hip_runtime.h>
#include <hip/hip_bf16.h>

typedef __attribute__((ext_vector_type(16))) _Float16 v16h;
typedef __attribute__((ext_vector_type(8)))  _Float16 v8h;
typedef __attribute__((ext_vector_type(8)))  float    v8f;

#define D_MODEL 512
#define N_HEADS 8
#define HEAD_DIM 64
#define SEQ 2048
#define BATCH 8
#define ROWS (BATCH * SEQ)   /* 16384 */

// ---------------------------------------------------------------------------
// A/B tile loader for v_wmma_f32_16x16x32_f16.
// Loads a 16x32 f16 tile from a row-major matrix (row stride `ld` halves)
// in the A-operand lane layout (identical to the B-operand layout when the
// source is B^T stored row-major, i.e. torch Linear weights [N,K]).
//   lane<16 : row = row0+lane,    K = k0+{0..7, 16..23}
//   lane>=16: row = row0+lane-16, K = k0+{8..15, 24..31}
// ---------------------------------------------------------------------------
__device__ __forceinline__ v16h load_ab_tile(const _Float16* __restrict__ base,
                                             int row0, int k0, int ld, int lane) {
    const int r = row0 + (lane & 15);
    const int k = k0 + ((lane & 16) ? 8 : 0);
    const _Float16* p = base + (size_t)r * (size_t)ld + k;
    v8h lo = *(const v8h*)(p);
    v8h hi = *(const v8h*)(p + 16);
    return __builtin_shufflevector(lo, hi, 0, 1, 2, 3, 4, 5, 6, 7,
                                   8, 9, 10, 11, 12, 13, 14, 15);
}

__device__ __forceinline__ v8f wmma_f16(v16h a, v16h b, v8f c) {
    return __builtin_amdgcn_wmma_f32_16x16x32_f16(false, a, false, b,
                                                  (short)0, c, false, false);
}

__device__ __forceinline__ float hgrp_max(float v) {   // reduce across a 16-lane half
    v = fmaxf(v, __shfl_xor(v, 1, 32));
    v = fmaxf(v, __shfl_xor(v, 2, 32));
    v = fmaxf(v, __shfl_xor(v, 4, 32));
    v = fmaxf(v, __shfl_xor(v, 8, 32));
    return v;
}
__device__ __forceinline__ float hgrp_sum(float v) {
    v += __shfl_xor(v, 1, 32);
    v += __shfl_xor(v, 2, 32);
    v += __shfl_xor(v, 4, 32);
    v += __shfl_xor(v, 8, 32);
    return v;
}

// ---------------------------------------------------------------------------
// Kernel 1: LayerNorm (eps=1e-5) + cast to f16.  One wave per row of 512.
// ---------------------------------------------------------------------------
__global__ void __launch_bounds__(256)
ln_kernel(const float* __restrict__ x, const float* __restrict__ gamma,
          const float* __restrict__ beta, _Float16* __restrict__ xh) {
    const int lane = threadIdx.x & 31;
    const int wave = threadIdx.x >> 5;
    const int row  = blockIdx.x * 8 + wave;

    const float* xr = x + (size_t)row * D_MODEL + lane * 16;
    float v[16];
    {
        float4 d0 = ((const float4*)xr)[0];
        float4 d1 = ((const float4*)xr)[1];
        float4 d2 = ((const float4*)xr)[2];
        float4 d3 = ((const float4*)xr)[3];
        v[0]=d0.x; v[1]=d0.y; v[2]=d0.z; v[3]=d0.w;
        v[4]=d1.x; v[5]=d1.y; v[6]=d1.z; v[7]=d1.w;
        v[8]=d2.x; v[9]=d2.y; v[10]=d2.z; v[11]=d2.w;
        v[12]=d3.x; v[13]=d3.y; v[14]=d3.z; v[15]=d3.w;
    }
    float sum = 0.f, sq = 0.f;
#pragma unroll
    for (int j = 0; j < 16; ++j) { sum += v[j]; sq += v[j] * v[j]; }
#pragma unroll
    for (int m = 16; m >= 1; m >>= 1) {
        sum += __shfl_xor(sum, m, 32);
        sq  += __shfl_xor(sq,  m, 32);
    }
    const float mean = sum * (1.f / (float)D_MODEL);
    const float var  = sq * (1.f / (float)D_MODEL) - mean * mean;
    const float rstd = rsqrtf(var + 1e-5f);

    const float* g = gamma + lane * 16;
    const float* bt = beta + lane * 16;
    _Float16 o[16];
#pragma unroll
    for (int j = 0; j < 16; ++j)
        o[j] = (_Float16)((v[j] - mean) * rstd * g[j] + bt[j]);

    v8h* op = (v8h*)(xh + (size_t)row * D_MODEL + lane * 16);
    op[0] = *(v8h*)&o[0];
    op[1] = *(v8h*)&o[8];
}

// ---------------------------------------------------------------------------
// Kernel 2: f32 -> f16 elementwise (weights)
// ---------------------------------------------------------------------------
__global__ void cvt_kernel(const float* __restrict__ in, _Float16* __restrict__ out, int n) {
    int i = blockIdx.x * blockDim.x + threadIdx.x;
    if (i < n) out[i] = (_Float16)in[i];
}

// ---------------------------------------------------------------------------
// Kernel 3: QKV GEMM  [16384,512] x [1536,512]^T + bias, scatter to
//   Q [b,h,s,hd] (pre-scaled by 1/8),  K [b,h,s,hd],  V^T [b,h,hd,s]  (f16)
// Wave computes a 16x64 tile; K-loop is software-pipelined (double-buffered
// tile registers) so WMMAs consume tiles loaded one iteration earlier.
// ---------------------------------------------------------------------------
__global__ void __launch_bounds__(256)
qkv_gemm_kernel(const _Float16* __restrict__ A, const _Float16* __restrict__ Bt,
                const float* __restrict__ bias,
                _Float16* __restrict__ Qd, _Float16* __restrict__ Kd,
                _Float16* __restrict__ Vt) {
    const int lane = threadIdx.x & 31;
    const int wave = threadIdx.x >> 5;
    const int m0 = blockIdx.x * 16;
    const int n0 = (blockIdx.y * 8 + wave) * 64;

    v8f acc[4];
#pragma unroll
    for (int nt = 0; nt < 4; ++nt)
        acc[nt] = (v8f){0.f, 0.f, 0.f, 0.f, 0.f, 0.f, 0.f, 0.f};

    // prologue loads
    v16h a  = load_ab_tile(A,  m0, 0, D_MODEL, lane);
    v16h b0 = load_ab_tile(Bt, n0,      0, D_MODEL, lane);
    v16h b1 = load_ab_tile(Bt, n0 + 16, 0, D_MODEL, lane);
    v16h b2 = load_ab_tile(Bt, n0 + 32, 0, D_MODEL, lane);
    v16h b3 = load_ab_tile(Bt, n0 + 48, 0, D_MODEL, lane);

#pragma unroll 3
    for (int k0 = 0; k0 < D_MODEL - 32; k0 += 32) {
        const int kn = k0 + 32;
        // prefetch next K-step into fresh registers
        const v16h an  = load_ab_tile(A,  m0, kn, D_MODEL, lane);
        const v16h c0  = load_ab_tile(Bt, n0,      kn, D_MODEL, lane);
        const v16h c1  = load_ab_tile(Bt, n0 + 16, kn, D_MODEL, lane);
        const v16h c2  = load_ab_tile(Bt, n0 + 32, kn, D_MODEL, lane);
        const v16h c3  = load_ab_tile(Bt, n0 + 48, kn, D_MODEL, lane);
        // consume current tiles (loaded last iteration -> partial waits)
        acc[0] = wmma_f16(a, b0, acc[0]);
        acc[1] = wmma_f16(a, b1, acc[1]);
        acc[2] = wmma_f16(a, b2, acc[2]);
        acc[3] = wmma_f16(a, b3, acc[3]);
        a = an; b0 = c0; b1 = c1; b2 = c2; b3 = c3;
    }
    acc[0] = wmma_f16(a, b0, acc[0]);
    acc[1] = wmma_f16(a, b1, acc[1]);
    acc[2] = wmma_f16(a, b2, acc[2]);
    acc[3] = wmma_f16(a, b3, acc[3]);

    const int col   = lane & 15;
    const int hi8   = (lane & 16) ? 8 : 0;
    const int which = n0 >> 9;            // 0=Q 1=K 2=V (uniform for tile)
    const int head  = (n0 & 511) >> 6;    // uniform for tile

#pragma unroll
    for (int nt = 0; nt < 4; ++nt) {
        const int d = nt * 16 + col;
        const float bn = bias[n0 + nt * 16 + col];
#pragma unroll
        for (int v = 0; v < 8; ++v) {
            const int m = m0 + v + hi8;
            const int b = m >> 11;        // /2048
            const int s = m & 2047;
            const float val = acc[nt][v] + bn;
            const size_t bh = (size_t)(b * N_HEADS + head);
            if (which == 0) {
                Qd[(bh * SEQ + s) * HEAD_DIM + d] = (_Float16)(val * 0.125f); // 1/sqrt(64)
            } else if (which == 1) {
                Kd[(bh * SEQ + s) * HEAD_DIM + d] = (_Float16)val;
            } else {
                Vt[(bh * HEAD_DIM + d) * SEQ + s] = (_Float16)val;
            }
        }
    }
}

// ---------------------------------------------------------------------------
// Kernel 4: flash attention.  Wave owns 16 queries of one (b,h); per 32-key
// step: all 8 B tiles (4 K + 4 V) are loaded up front (one clause burst),
// S-WMMAs consume the K tiles with partial waits while V-tile latency hides
// behind the softmax VALU chain and the LDS P-staging round-trip.
// ---------------------------------------------------------------------------
__global__ void __launch_bounds__(128)
attn_kernel(const _Float16* __restrict__ Q, const _Float16* __restrict__ Kd,
            const _Float16* __restrict__ Vt, _Float16* __restrict__ attn_out) {
    __shared__ _Float16 pbuf[4 * 16 * 32];

    const int lane = threadIdx.x & 31;
    const int wave = threadIdx.x >> 5;
    const int head = blockIdx.y;
    const int b    = blockIdx.z;
    const int bh   = b * N_HEADS + head;
    const int q0   = (blockIdx.x * 4 + wave) * 16;

    const _Float16* Qh = Q  + (size_t)bh * SEQ * HEAD_DIM;
    const _Float16* Kh = Kd + (size_t)bh * SEQ * HEAD_DIM;
    const _Float16* Vh = Vt + (size_t)bh * HEAD_DIM * SEQ;
    _Float16* pw = pbuf + wave * (16 * 32);

    const int col = lane & 15;
    const int hi8 = (lane & 16) ? 8 : 0;

    // Q tile resident in registers (16x64 = two 16x32 A operands)
    const v16h qa0 = load_ab_tile(Qh, q0, 0,  HEAD_DIM, lane);
    const v16h qa1 = load_ab_tile(Qh, q0, 32, HEAD_DIM, lane);

    float rowmax[8], rowsum[8];
    v8f o0 = {0.f,0.f,0.f,0.f,0.f,0.f,0.f,0.f};
    v8f o1 = o0, o2 = o0, o3 = o0;
#pragma unroll
    for (int v = 0; v < 8; ++v) { rowmax[v] = -1e30f; rowsum[v] = 0.f; }

    for (int kb = 0; kb < SEQ; kb += 32) {
        // ----- issue ALL B-tile loads for this step up front -----
        const v16h kt00 = load_ab_tile(Kh, kb,      0,  HEAD_DIM, lane);
        const v16h kt01 = load_ab_tile(Kh, kb,      32, HEAD_DIM, lane);
        const v16h kt10 = load_ab_tile(Kh, kb + 16, 0,  HEAD_DIM, lane);
        const v16h kt11 = load_ab_tile(Kh, kb + 16, 32, HEAD_DIM, lane);
        const v16h vt0  = load_ab_tile(Vh, 0,  kb, SEQ, lane);
        const v16h vt1  = load_ab_tile(Vh, 16, kb, SEQ, lane);
        const v16h vt2  = load_ab_tile(Vh, 32, kb, SEQ, lane);
        const v16h vt3  = load_ab_tile(Vh, 48, kb, SEQ, lane);

        // ----- scores S = Q K^T for 32 keys (two 16x16 C tiles) -----
        v8f s0 = {0.f,0.f,0.f,0.f,0.f,0.f,0.f,0.f};
        v8f s1 = s0;
        s0 = wmma_f16(qa0, kt00, s0);
        s0 = wmma_f16(qa1, kt01, s0);
        s1 = wmma_f16(qa0, kt10, s1);
        s1 = wmma_f16(qa1, kt11, s1);

        // ----- online softmax (row = across 16-lane half per C VGPR) -----
        float p0[8], p1[8], alpha[8];
#pragma unroll
        for (int v = 0; v < 8; ++v) {
            float mloc = hgrp_max(fmaxf(s0[v], s1[v]));
            float mnew = fmaxf(rowmax[v], mloc);
            alpha[v]  = __expf(rowmax[v] - mnew);
            rowmax[v] = mnew;
            p0[v] = __expf(s0[v] - mnew);
            p1[v] = __expf(s1[v] - mnew);
            rowsum[v] = rowsum[v] * alpha[v] + hgrp_sum(p0[v] + p1[v]);
        }
#pragma unroll
        for (int v = 0; v < 8; ++v) {
            o0[v] *= alpha[v]; o1[v] *= alpha[v];
            o2[v] *= alpha[v]; o3[v] *= alpha[v];
            // stage P (C layout -> row-major [16][32] in LDS)
            const int M = v + hi8;
            pw[M * 32 + col]      = (_Float16)p0[v];
            pw[M * 32 + 16 + col] = (_Float16)p1[v];
        }
        asm volatile("s_wait_dscnt 0" ::: "memory");   // LDS store->load fence (same wave, in-order HW)

        // ----- O += P @ V  (A = P 16x32 from LDS, V tiles already resident) -----
        const v16h pa = load_ab_tile(pw, 0, 0, 32, lane);
        o0 = wmma_f16(pa, vt0, o0);
        o1 = wmma_f16(pa, vt1, o1);
        o2 = wmma_f16(pa, vt2, o2);
        o3 = wmma_f16(pa, vt3, o3);
        asm volatile("" ::: "memory");
    }

    // ----- normalize and emit [b, s, h*hd] f16 for the out-projection -----
#pragma unroll
    for (int v = 0; v < 8; ++v) {
        const float inv = 1.0f / rowsum[v];
        const int M = v + hi8;
        const size_t off = (size_t)(b * SEQ + q0 + M) * D_MODEL
                         + head * HEAD_DIM + col;
        attn_out[off]      = (_Float16)(o0[v] * inv);
        attn_out[off + 16] = (_Float16)(o1[v] * inv);
        attn_out[off + 32] = (_Float16)(o2[v] * inv);
        attn_out[off + 48] = (_Float16)(o3[v] * inv);
    }
}

// ---------------------------------------------------------------------------
// Kernel 5: out projection [16384,512] x [512,512]^T + bias -> f32 output
// Wave computes a 16x64 tile; software-pipelined K loop.
// ---------------------------------------------------------------------------
__global__ void __launch_bounds__(256)
out_gemm_kernel(const _Float16* __restrict__ A, const _Float16* __restrict__ Bt,
                const float* __restrict__ bias, float* __restrict__ out) {
    const int lane = threadIdx.x & 31;
    const int wave = threadIdx.x >> 5;
    const int m0 = blockIdx.x * 16;
    const int n0 = wave * 64;            // 8 waves cover N=512

    v8f acc[4];
#pragma unroll
    for (int nt = 0; nt < 4; ++nt)
        acc[nt] = (v8f){0.f, 0.f, 0.f, 0.f, 0.f, 0.f, 0.f, 0.f};

    v16h a  = load_ab_tile(A,  m0, 0, D_MODEL, lane);
    v16h b0 = load_ab_tile(Bt, n0,      0, D_MODEL, lane);
    v16h b1 = load_ab_tile(Bt, n0 + 16, 0, D_MODEL, lane);
    v16h b2 = load_ab_tile(Bt, n0 + 32, 0, D_MODEL, lane);
    v16h b3 = load_ab_tile(Bt, n0 + 48, 0, D_MODEL, lane);

#pragma unroll 3
    for (int k0 = 0; k0 < D_MODEL - 32; k0 += 32) {
        const int kn = k0 + 32;
        const v16h an = load_ab_tile(A,  m0, kn, D_MODEL, lane);
        const v16h c0 = load_ab_tile(Bt, n0,      kn, D_MODEL, lane);
        const v16h c1 = load_ab_tile(Bt, n0 + 16, kn, D_MODEL, lane);
        const v16h c2 = load_ab_tile(Bt, n0 + 32, kn, D_MODEL, lane);
        const v16h c3 = load_ab_tile(Bt, n0 + 48, kn, D_MODEL, lane);
        acc[0] = wmma_f16(a, b0, acc[0]);
        acc[1] = wmma_f16(a, b1, acc[1]);
        acc[2] = wmma_f16(a, b2, acc[2]);
        acc[3] = wmma_f16(a, b3, acc[3]);
        a = an; b0 = c0; b1 = c1; b2 = c2; b3 = c3;
    }
    acc[0] = wmma_f16(a, b0, acc[0]);
    acc[1] = wmma_f16(a, b1, acc[1]);
    acc[2] = wmma_f16(a, b2, acc[2]);
    acc[3] = wmma_f16(a, b3, acc[3]);

    const int col = lane & 15;
    const int hi8 = (lane & 16) ? 8 : 0;
#pragma unroll
    for (int nt = 0; nt < 4; ++nt) {
        const int n = n0 + nt * 16 + col;
        const float bn = bias[n];
#pragma unroll
        for (int v = 0; v < 8; ++v)
            out[(size_t)(m0 + v + hi8) * D_MODEL + n] = acc[nt][v] + bn;
    }
}

// ---------------------------------------------------------------------------
extern "C" void kernel_launch(void* const* d_in, const int* in_sizes, int n_in,
                              void* d_out, int out_size, void* d_ws, size_t ws_size,
                              hipStream_t stream) {
    const float* x        = (const float*)d_in[0];
    const float* ln_gamma = (const float*)d_in[1];
    const float* ln_beta  = (const float*)d_in[2];
    const float* W_qkv    = (const float*)d_in[3];
    const float* b_qkv    = (const float*)d_in[4];
    const float* W_out    = (const float*)d_in[5];
    const float* b_out    = (const float*)d_in[6];
    float* out = (float*)d_out;

    char* ws = (char*)d_ws;
    const size_t SZ_ACT = (size_t)ROWS * D_MODEL * sizeof(_Float16);          // 16 MB
    const size_t SZ_WQ  = (size_t)3 * D_MODEL * D_MODEL * sizeof(_Float16);   // 1.5 MB
    const size_t SZ_WO  = (size_t)D_MODEL * D_MODEL * sizeof(_Float16);       // 0.5 MB

    _Float16* xh    = (_Float16*)(ws);
    _Float16* Wq16  = (_Float16*)(ws + SZ_ACT);
    _Float16* Wo16  = (_Float16*)(ws + SZ_ACT + SZ_WQ);
    _Float16* Qd    = (_Float16*)(ws + SZ_ACT + SZ_WQ + SZ_WO);
    _Float16* Kd    = (_Float16*)(ws + 2 * SZ_ACT + SZ_WQ + SZ_WO);
    _Float16* Vt    = (_Float16*)(ws + 3 * SZ_ACT + SZ_WQ + SZ_WO);
    _Float16* attnb = (_Float16*)(ws + 4 * SZ_ACT + SZ_WQ + SZ_WO);

    // 1) LayerNorm -> f16
    ln_kernel<<<ROWS / 8, 256, 0, stream>>>(x, ln_gamma, ln_beta, xh);
    // 2) weights -> f16
    cvt_kernel<<<(3 * D_MODEL * D_MODEL + 255) / 256, 256, 0, stream>>>(W_qkv, Wq16, 3 * D_MODEL * D_MODEL);
    cvt_kernel<<<(D_MODEL * D_MODEL + 255) / 256, 256, 0, stream>>>(W_out, Wo16, D_MODEL * D_MODEL);
    // 3) QKV projection + scatter  (N super-tiles of 64: 1536/64 = 24 = 3 blocks * 8 waves)
    qkv_gemm_kernel<<<dim3(ROWS / 16, 3), 256, 0, stream>>>(xh, Wq16, b_qkv, Qd, Kd, Vt);
    // 4) flash attention
    attn_kernel<<<dim3(SEQ / (16 * 4), N_HEADS, BATCH), 128, 0, stream>>>(Qd, Kd, Vt, attnb);
    // 5) output projection (8 waves cover N=512)
    out_gemm_kernel<<<dim3(ROWS / 16, 1), 256, 0, stream>>>(attnb, Wo16, b_out, out);
}
